// BitoGuardGNN_42399917146564
// MI455X (gfx1250) — matrix operand order repaired and checked by reading, not compiled, not run on hardware
//
#include <hip/hip_runtime.h>
#include <stdint.h>

// ---------------------------------------------------------------------------
// Types for CDNA5 WMMA (wave32)
// ---------------------------------------------------------------------------
typedef __attribute__((ext_vector_type(8)))  float    v8f;
typedef __attribute__((ext_vector_type(16))) __bf16   v16bf;
typedef __attribute__((ext_vector_type(4)))  unsigned uint4v;

union Frag16 { v16bf v; uint4v q[2]; };

__device__ __forceinline__ unsigned short f2bf(float f) {
  unsigned u = __float_as_uint(f);
  u += 0x7FFFu + ((u >> 16) & 1u);            // round-to-nearest-even
  return (unsigned short)(u >> 16);
}
// monotonic float -> unsigned key (for atomicMax over floats)
__device__ __forceinline__ unsigned fkey(float f) {
  unsigned u = __float_as_uint(f);
  return (u & 0x80000000u) ? ~u : (u | 0x80000000u);
}
__device__ __forceinline__ float fdecode(unsigned k) {
  unsigned u = (k & 0x80000000u) ? (k & 0x7FFFFFFFu) : ~k;
  return __uint_as_float(u);
}

// CDNA5 async global->LDS copy (ASYNCcnt-tracked), 16B per lane per issue.
// lds_off is a byte offset into the workgroup LDS segment (dynamic LDS base 0).
__device__ __forceinline__ void async_load_b128(unsigned lds_off, const void* gptr) {
  asm volatile("global_load_async_to_lds_b128 %0, %1, off"
               :: "v"(lds_off), "v"((unsigned long long)(size_t)gptr)
               : "memory");
}
__device__ __forceinline__ void wait_async0() {
  asm volatile("s_wait_asynccnt 0" ::: "memory");
}
__device__ __forceinline__ void wait_async2() {  // allow 2 newest (next buffer) in flight
  asm volatile("s_wait_asynccnt 2" ::: "memory");
}

// ---------------------------------------------------------------------------
// Fragment load (two 16B LDS chunks each, contiguous k = 16*lg + s in A and B) + MMA
// LDS tile layout: rows of 40 ushorts (80 B stride, 16B-aligned chunks).
// ---------------------------------------------------------------------------
__device__ __forceinline__ void tile_mma(const unsigned char* lA, const unsigned char* lB,
                                         int ar, int cn, int lm, int lg,
                                         v8f& acc0, v8f& acc1)
{
  Frag16 fa, fb0, fb1;
  const unsigned char* pa = lA + ar * 80 + lg * 32;          // A 16x32 MxK
  fa.q[0] = *(const uint4v*)pa;
  fa.q[1] = *(const uint4v*)(pa + 16);
  const unsigned char* pb0 = lB + (cn + lm) * 80 + lg * 32;  // B 32x16 KxN
  fb0.q[0] = *(const uint4v*)pb0;
  fb0.q[1] = *(const uint4v*)(pb0 + 16);
  const unsigned char* pb1 = lB + (cn + 16 + lm) * 80 + lg * 32;
  fb1.q[0] = *(const uint4v*)pb1;
  fb1.q[1] = *(const uint4v*)(pb1 + 16);
  acc0 = __builtin_amdgcn_wmma_f32_16x16x32_bf16(
      false, fa.v, false, fb0.v, (short)0, acc0, false, false);
  acc1 = __builtin_amdgcn_wmma_f32_16x16x32_bf16(
      false, fa.v, false, fb1.v, (short)0, acc1, false, false);
}

// ---------------------------------------------------------------------------
// GEMM: C[M,N] = A[M,K](bf16,row-major) @ W[K,N] + bias[N]
// Weights supplied pre-transposed: BT[N,K] bf16 -> both LDS tiles K-contiguous.
// Block tile 64x64, 8 waves; wave w -> rows 16*(w&3), cols 32*(w>>2)..+31.
// Interior blocks: double-buffered async global->LDS pipeline.
// Dynamic LDS: A0@0, B0@5120, A1@10240, B1@15360 (each 64 rows x 80 B).
// ---------------------------------------------------------------------------
#define GEMM_LDS_BYTES 20480

__global__ __launch_bounds__(256)
void gemm_bf16_wmma(const unsigned short* __restrict__ A,
                    const unsigned short* __restrict__ BT,
                    const float* __restrict__ bias,
                    float* __restrict__ Cf,            // f32 out (or null)
                    unsigned short* __restrict__ Cbf,  // bf16 out (or null)
                    int M, int N, int K)
{
  extern __shared__ __align__(16) unsigned char smem[];

  const int tid  = threadIdx.x;
  const int wave = tid >> 5;
  const int lane = tid & 31;
  const int m0 = blockIdx.y * 64;
  const int n0 = blockIdx.x * 64;
  const int rm = (wave & 3) * 16;     // wave row tile inside block
  const int cn = (wave >> 2) * 32;    // wave col base (covers cn..cn+31)
  const int lm = lane & 15;
  const int lg = lane >> 4;
  const int ar = rm + lm;

  const bool interior = (m0 + 64 <= M) && ((K & 31) == 0);
  const int srow = tid >> 2;          // 0..63 (tile row this thread stages)
  const int sq   = tid & 3;           // 16-byte chunk within the 64B row
  const unsigned sOff = (unsigned)(srow * 80 + sq * 16);
  const unsigned short* gA = A  + (size_t)(m0 + srow) * K + sq * 8;
  const unsigned short* gB = BT + (size_t)(n0 + srow) * K + sq * 8;

  v8f acc0 = {};
  v8f acc1 = {};

  if (interior) {
    // prologue: stage K-tile 0 into buffer 0
    async_load_b128(sOff,         gA);
    async_load_b128(sOff + 5120u, gB);
    for (int kb = 0; kb < K; kb += 32) {
      unsigned cur = ((unsigned)(kb >> 5) & 1u) * 10240u;
      if (kb + 32 < K) {                        // prefetch next tile
        unsigned nxt = 10240u - cur;
        async_load_b128(sOff + nxt,         gA + kb + 32);
        async_load_b128(sOff + nxt + 5120u, gB + kb + 32);
        wait_async2();                          // drain current-buffer copies
      } else {
        wait_async0();
      }
      __syncthreads();
      tile_mma(smem + cur, smem + cur + 5120, ar, cn, lm, lg, acc0, acc1);
      __syncthreads();                          // reads done before next overwrite
    }
  } else {
    unsigned short* lA = (unsigned short*)smem;
    unsigned short* lB = (unsigned short*)(smem + 5120);
    for (int kb = 0; kb < K; kb += 32) {
      for (int i = tid; i < 1024; i += 256) {   // A: 64x32 as dwords
        int row = i >> 4, c2 = i & 15;
        int m = m0 + row, k = kb + c2 * 2;
        unsigned v = 0;
        if (m < M && k < K) v = *(const unsigned*)(A + (size_t)m * K + k);
        *(unsigned*)(lA + row * 40 + c2 * 2) = v;
      }
      for (int i = tid; i < 1024; i += 256) {   // B: 64x32 as dwords
        int row = i >> 4, c2 = i & 15;
        int k = kb + c2 * 2;
        unsigned v = 0;
        if (k < K) v = *(const unsigned*)(BT + (size_t)(n0 + row) * K + k);
        *(unsigned*)(lB + row * 40 + c2 * 2) = v;
      }
      __syncthreads();
      tile_mma(smem, smem + 5120, ar, cn, lm, lg, acc0, acc1);
      __syncthreads();
    }
  }

  // -- epilogue: D layout = VGPR r: lanes0-15 -> M=r, lanes16-31 -> M=8+r
#pragma unroll
  for (int r = 0; r < 8; ++r) {
    int m = m0 + rm + (lg ? r + 8 : r);
    if (m >= M) continue;
    int n = n0 + cn + lm;
    float v0 = acc0[r] + bias[n];
    float v1 = acc1[r] + bias[n + 16];
    if (Cf)  { Cf[(size_t)m * N + n]  = v0;      Cf[(size_t)m * N + n + 16]  = v1; }
    if (Cbf) { Cbf[(size_t)m * N + n] = f2bf(v0); Cbf[(size_t)m * N + n + 16] = f2bf(v1); }
  }
}

// ---------------------------------------------------------------------------
// Edge phase (GATv2).  One wave32 per edge; lane owns channels 4L..4L+3 so
// every gather is a 16B global_load_b128 and scatters are contiguous atomics.
// ---------------------------------------------------------------------------
__global__ __launch_bounds__(256)
void edge_logits_max(const float* __restrict__ xl, const float* __restrict__ xr,
                     const float* __restrict__ ea, const int* __restrict__ src,
                     const int* __restrict__ dst, const float* __restrict__ att,
                     const float* __restrict__ We, float* __restrict__ logit,
                     unsigned* __restrict__ segkey, int E, int H)
{
  int e = blockIdx.x * 8 + (threadIdx.x >> 5);
  int lane = threadIdx.x & 31;
  if (e >= E) return;
  int s = src[e], d = dst[e];
  float a = ea[e];
  int HC = H * 128;
  const float4* xls = (const float4*)(xl + (size_t)s * HC);
  const float4* xrd = (const float4*)(xr + (size_t)d * HC);
  const float4* We4 = (const float4*)We;
  const float4* at4 = (const float4*)att;
  for (int h = 0; h < H; ++h) {
    int q = h * 32 + lane;
    float4 l = xls[q], r = xrd[q], w = We4[q], t = at4[q];
    float zx = l.x + r.x + a * w.x; zx = (zx > 0.f) ? zx : 0.2f * zx;
    float zy = l.y + r.y + a * w.y; zy = (zy > 0.f) ? zy : 0.2f * zy;
    float zz = l.z + r.z + a * w.z; zz = (zz > 0.f) ? zz : 0.2f * zz;
    float zw = l.w + r.w + a * w.w; zw = (zw > 0.f) ? zw : 0.2f * zw;
    float sum = t.x * zx + t.y * zy + t.z * zz + t.w * zw;
#pragma unroll
    for (int off = 16; off > 0; off >>= 1) sum += __shfl_down(sum, off, 32);
    if (lane == 0) {
      logit[(size_t)e * H + h] = sum;
      atomicMax(&segkey[(size_t)d * H + h], fkey(sum));
    }
  }
}

__global__ void edge_exp_sum(const int* __restrict__ dst, float* __restrict__ ex,
                             const unsigned* __restrict__ segkey,
                             float* __restrict__ segsum, int E, int H)
{
  int i = blockIdx.x * blockDim.x + threadIdx.x;
  if (i >= E * H) return;
  int e = i / H, h = i - e * H;
  int d = dst[e];
  float m = fdecode(segkey[(size_t)d * H + h]);
  float x = expf(ex[i] - m);                   // ex holds logits on entry
  ex[i] = x;
  atomicAdd(&segsum[(size_t)d * H + h], x);
}

__global__ __launch_bounds__(256)
void edge_aggregate(const float* __restrict__ xl, const float* __restrict__ ex,
                    const float* __restrict__ segsum, const int* __restrict__ src,
                    const int* __restrict__ dst, float* __restrict__ acc, int E, int H)
{
  int e = blockIdx.x * 8 + (threadIdx.x >> 5);
  int lane = threadIdx.x & 31;
  if (e >= E) return;
  int s = src[e], d = dst[e];
  int HC = H * 128;
  const float4* xls = (const float4*)(xl + (size_t)s * HC);
  float* accd = acc + (size_t)d * HC;
  for (int h = 0; h < H; ++h) {
    float alpha = ex[(size_t)e * H + h] / (segsum[(size_t)d * H + h] + 1e-16f);
    float4 l = xls[h * 32 + lane];
    float* ad = accd + h * 128 + lane * 4;
    atomicAdd(ad + 0, l.x * alpha);
    atomicAdd(ad + 1, l.y * alpha);
    atomicAdd(ad + 2, l.z * alpha);
    atomicAdd(ad + 3, l.w * alpha);
  }
}

// ---------------------------------------------------------------------------
// Elementwise helpers
// ---------------------------------------------------------------------------
__global__ void k_fill_u32(unsigned* p, unsigned v, int n) {
  int i = blockIdx.x * blockDim.x + threadIdx.x;
  if (i < n) p[i] = v;
}
__global__ void k_fill_f32(float* p, float v, int n) {
  int i = blockIdx.x * blockDim.x + threadIdx.x;
  if (i < n) p[i] = v;
}
__global__ void k_f32_to_bf16(const float* __restrict__ a, unsigned short* __restrict__ b, int n) {
  int i = blockIdx.x * blockDim.x + threadIdx.x;
  if (i < n) b[i] = f2bf(a[i]);
}
// transpose-convert: W[K,N] f32 -> WT[N,K] bf16
__global__ void k_f32_to_bf16_T(const float* __restrict__ W, unsigned short* __restrict__ WT,
                                int K, int N) {
  int i = blockIdx.x * blockDim.x + threadIdx.x;
  if (i >= K * N) return;
  int k = i / N, n = i - k * N;
  WT[(size_t)n * K + k] = f2bf(W[i]);
}
__global__ void k_relu_to_bf16(const float* __restrict__ a, unsigned short* __restrict__ b, int n) {
  int i = blockIdx.x * blockDim.x + threadIdx.x;
  if (i < n) { float v = a[i]; b[i] = f2bf(v > 0.f ? v : 0.f); }
}
__global__ void k_init_bias(float* __restrict__ acc, const float* __restrict__ b1,
                            const float* __restrict__ b2, int rows, int cols) {
  int i = blockIdx.x * blockDim.x + threadIdx.x;
  if (i >= rows * cols) return;
  int c = i % cols;
  acc[i] = b1[c] + (b2 ? b2[c] : 0.f);
}

// ---------------------------------------------------------------------------
// Final: out[n,:] = (relu(u2[n,:]) * sigmoid(cate[n]*Wg + bg)) @ Wo + bo
// ---------------------------------------------------------------------------
__global__ __launch_bounds__(64)
void final_kernel(const float* __restrict__ u2, const float* __restrict__ cate,
                  const float* __restrict__ Wg, const float* __restrict__ bg,
                  const float* __restrict__ Wo, const float* __restrict__ bo,
                  float* __restrict__ out)
{
  __shared__ float v[128];
  int n = blockIdx.x;
  int t = threadIdx.x;
  float cs = cate[n];
  for (int c = t; c < 128; c += 64) {
    float g = 1.f / (1.f + expf(-(cs * Wg[c] + bg[c])));
    float u = u2[(size_t)n * 128 + c];
    v[c] = (u > 0.f ? u : 0.f) * g;
  }
  __syncthreads();
  float acc = bo[t];
#pragma unroll 4
  for (int c = 0; c < 128; ++c) acc += v[c] * Wo[c * 64 + t];
  out[(size_t)n * 64 + t] = acc;
}

// ---------------------------------------------------------------------------
// Host orchestration
// ---------------------------------------------------------------------------
static inline void launch_gemm(const unsigned short* A, const unsigned short* BT,
                               const float* bias, float* Cf, unsigned short* Cbf,
                               int M, int N, int K, hipStream_t s)
{
  dim3 grid(N / 64, (M + 63) / 64);
  gemm_bf16_wmma<<<grid, dim3(256), GEMM_LDS_BYTES, s>>>(A, BT, bias, Cf, Cbf, M, N, K);
}
static inline void conv_bf16(const float* a, unsigned short* b, int n, hipStream_t s) {
  k_f32_to_bf16<<<(n + 255) / 256, 256, 0, s>>>(a, b, n);
}
static inline void conv_bf16_T(const float* W, unsigned short* WT, int K, int N, hipStream_t s) {
  int n = K * N;
  k_f32_to_bf16_T<<<(n + 255) / 256, 256, 0, s>>>(W, WT, K, N);
}
static inline void run_edges(const float* xl, const float* xr, const float* ea,
                             const int* src, const int* dst, const float* att,
                             const float* We, float* exbuf, unsigned* segk,
                             float* segs, float* acc, int E, int Nd, int H,
                             hipStream_t s)
{
  int nk = Nd * H;
  k_fill_u32<<<(nk + 255) / 256, 256, 0, s>>>(segk, 0u, nk);
  k_fill_f32<<<(nk + 255) / 256, 256, 0, s>>>(segs, 0.f, nk);
  int gb = (E + 7) / 8;
  edge_logits_max<<<gb, 256, 0, s>>>(xl, xr, ea, src, dst, att, We, exbuf, segk, E, H);
  int t = E * H;
  edge_exp_sum<<<(t + 255) / 256, 256, 0, s>>>(dst, exbuf, segk, segs, E, H);
  edge_aggregate<<<gb, 256, 0, s>>>(xl, exbuf, segs, src, dst, acc, E, H);
}

extern "C" void kernel_launch(void* const* d_in, const int* in_sizes, int n_in,
                              void* d_out, int out_size, void* d_ws, size_t ws_size,
                              hipStream_t stream)
{
  const int NU = in_sizes[0] / 32;
  const int NW = in_sizes[1] / 16;
  const int E  = in_sizes[2] / 2;
  const int Nmax = NU > NW ? NU : NW;

  const float* x_user    = (const float*)d_in[0];
  const float* x_wallet  = (const float*)d_in[1];
  const int*   ei_uw     = (const int*)d_in[2];
  const int*   ei_wu     = (const int*)d_in[3];
  const int*   ei_uu     = (const int*)d_in[4];
  const float* ea_uw     = (const float*)d_in[5];
  const float* ea_wu     = (const float*)d_in[6];
  const float* ea_uu     = (const float*)d_in[7];
  const float* cate      = (const float*)d_in[8];
  const float* Wp_user   = (const float*)d_in[9];
  const float* bp_user   = (const float*)d_in[10];
  const float* Wp_wallet = (const float*)d_in[11];
  const float* bp_wallet = (const float*)d_in[12];
  // conv param dicts flatten as: Wl, bl, Wr, br, We, att, b
  const int B_UW1 = 13, B_WU1 = 20, B_UU1 = 27, /*B_UW2=34 unused,*/ B_WU2 = 41, B_UU2 = 48;
  const float* Wg = (const float*)d_in[55];
  const float* bg = (const float*)d_in[56];
  const float* Wo = (const float*)d_in[57];
  const float* bo = (const float*)d_in[58];
  auto P = [&](int base, int j) -> const float* { return (const float*)d_in[base + j]; };

  // ---- workspace arena ----
  char* base = (char*)d_ws;
  size_t off = 0;
  auto take = [&](size_t bytes) -> char* {
    char* p = base + off;
    off = (off + bytes + 255) & ~(size_t)255;
    return p;
  };
  unsigned short* xu_bf = (unsigned short*)take((size_t)NU * 32 * 2);
  unsigned short* xw_bf = (unsigned short*)take((size_t)NW * 16 * 2);
  unsigned short* hu_bf = (unsigned short*)take((size_t)NU * 128 * 2);
  unsigned short* hw_bf = (unsigned short*)take((size_t)NW * 128 * 2);
  unsigned short* w_pu  = (unsigned short*)take(32 * 128 * 2);
  unsigned short* w_pw  = (unsigned short*)take(16 * 128 * 2);
  unsigned short *wl1[3], *wr1[3], *wl2[2], *wr2[2];
  for (int i = 0; i < 3; ++i) { wl1[i] = (unsigned short*)take(128 * 512 * 2);
                                wr1[i] = (unsigned short*)take(128 * 512 * 2); }
  for (int i = 0; i < 2; ++i) { wl2[i] = (unsigned short*)take(512 * 128 * 2);
                                wr2[i] = (unsigned short*)take(512 * 128 * 2); }
  char* XL  = take((size_t)Nmax * 512 * 4);
  char* XR  = take((size_t)Nmax * 512 * 4);
  float* ACCU = (float*)take((size_t)NU * 512 * 4);
  float* ACCW = (float*)take((size_t)NW * 512 * 4);
  float* EX   = (float*)take((size_t)E * 4 * 4);
  unsigned* SEGK = (unsigned*)take((size_t)Nmax * 4 * 4);
  float*    SEGS = (float*)take((size_t)Nmax * 4 * 4);
  if (off > ws_size) return;  // workspace too small; nothing safe to do

  float* XLf = (float*)XL;
  float* XRf = (float*)XR;
  // layer-2 aliases inside freed layer-1 xl/xr regions
  unsigned short* hu2_bf = (unsigned short*)XL;                        // Nmax*512 bf16
  float* XL2f = (float*)(XL + (size_t)Nmax * 512 * 2);                 // Nmax*128 f32
  float* XR2f = XL2f + (size_t)Nmax * 128;                             // Nmax*128 f32
  unsigned short* hw2_bf = (unsigned short*)XR;                        // Nmax*512 bf16
  float* ACC2 = (float*)(XR + (size_t)Nmax * 512 * 2);                 // NU*128 f32

  // ---- convert inputs (row-major) + weights (transposed [N,K]) to bf16 ----
  conv_bf16(x_user, xu_bf, NU * 32, stream);
  conv_bf16(x_wallet, xw_bf, NW * 16, stream);
  conv_bf16_T(Wp_user, w_pu, 32, 128, stream);
  conv_bf16_T(Wp_wallet, w_pw, 16, 128, stream);
  const int b1[3] = { B_UW1, B_WU1, B_UU1 };
  for (int i = 0; i < 3; ++i) {
    conv_bf16_T(P(b1[i], 0), wl1[i], 128, 512, stream);
    conv_bf16_T(P(b1[i], 2), wr1[i], 128, 512, stream);
  }
  const int b2[2] = { B_WU2, B_UU2 };
  for (int i = 0; i < 2; ++i) {
    conv_bf16_T(P(b2[i], 0), wl2[i], 512, 128, stream);
    conv_bf16_T(P(b2[i], 2), wr2[i], 512, 128, stream);
  }

  // ---- input projections (WMMA) ----
  launch_gemm(xu_bf, w_pu, bp_user,   nullptr, hu_bf, NU, 128, 32, stream);
  launch_gemm(xw_bf, w_pw, bp_wallet, nullptr, hw_bf, NW, 128, 16, stream);

  // ---- layer 1, conv user->wallet (dst = wallet) ----
  launch_gemm(hu_bf, wl1[0], P(B_UW1, 1), XLf, nullptr, NU, 512, 128, stream);
  launch_gemm(hw_bf, wr1[0], P(B_UW1, 3), XRf, nullptr, NW, 512, 128, stream);
  k_init_bias<<<((NW * 512) + 255) / 256, 256, 0, stream>>>(ACCW, P(B_UW1, 6), nullptr, NW, 512);
  run_edges(XLf, XRf, ea_uw, ei_uw, ei_uw + E, P(B_UW1, 5), P(B_UW1, 4),
            EX, SEGK, SEGS, ACCW, E, NW, 4, stream);

  // ---- layer 1, conv wallet->user ----
  launch_gemm(hw_bf, wl1[1], P(B_WU1, 1), XLf, nullptr, NW, 512, 128, stream);
  launch_gemm(hu_bf, wr1[1], P(B_WU1, 3), XRf, nullptr, NU, 512, 128, stream);
  k_init_bias<<<((NU * 512) + 255) / 256, 256, 0, stream>>>(ACCU, P(B_WU1, 6), P(B_UU1, 6), NU, 512);
  run_edges(XLf, XRf, ea_wu, ei_wu, ei_wu + E, P(B_WU1, 5), P(B_WU1, 4),
            EX, SEGK, SEGS, ACCU, E, NU, 4, stream);

  // ---- layer 1, conv user->user (accumulates into ACCU) ----
  launch_gemm(hu_bf, wl1[2], P(B_UU1, 1), XLf, nullptr, NU, 512, 128, stream);
  launch_gemm(hu_bf, wr1[2], P(B_UU1, 3), XRf, nullptr, NU, 512, 128, stream);
  run_edges(XLf, XRf, ea_uu, ei_uu, ei_uu + E, P(B_UU1, 5), P(B_UU1, 4),
            EX, SEGK, SEGS, ACCU, E, NU, 4, stream);

  // ---- relu + requantize to bf16 for layer 2 ----
  k_relu_to_bf16<<<((NU * 512) + 255) / 256, 256, 0, stream>>>(ACCU, hu2_bf, NU * 512);
  k_relu_to_bf16<<<((NW * 512) + 255) / 256, 256, 0, stream>>>(ACCW, hw2_bf, NW * 512);

  // ---- layer 2, conv wallet->user (H=1) ----
  launch_gemm(hw2_bf, wl2[0], P(B_WU2, 1), XL2f, nullptr, NW, 128, 512, stream);
  launch_gemm(hu2_bf, wr2[0], P(B_WU2, 3), XR2f, nullptr, NU, 128, 512, stream);
  k_init_bias<<<((NU * 128) + 255) / 256, 256, 0, stream>>>(ACC2, P(B_WU2, 6), P(B_UU2, 6), NU, 128);
  run_edges(XL2f, XR2f, ea_wu, ei_wu, ei_wu + E, P(B_WU2, 5), P(B_WU2, 4),
            EX, SEGK, SEGS, ACC2, E, NU, 1, stream);

  // ---- layer 2, conv user->user (H=1, accumulates) ----
  launch_gemm(hu2_bf, wl2[1], P(B_UU2, 1), XL2f, nullptr, NU, 128, 512, stream);
  launch_gemm(hu2_bf, wr2[1], P(B_UU2, 3), XR2f, nullptr, NU, 128, 512, stream);
  run_edges(XL2f, XR2f, ea_uu, ei_uu, ei_uu + E, P(B_UU2, 5), P(B_UU2, 4),
            EX, SEGK, SEGS, ACC2, E, NU, 1, stream);

  // ---- gate + output projection ----
  final_kernel<<<NU, 64, 0, stream>>>(ACC2, cate, Wg, bg, Wo, bo, (float*)d_out);
}
